// SimpleGQA_24300924961357
// MI455X (gfx1250) — matrix operand order repaired and checked by reading, not compiled, not run on hardware
//
#include <hip/hip_runtime.h>
#include <stdint.h>

typedef __attribute__((ext_vector_type(16))) _Float16 v16h;
typedef __attribute__((ext_vector_type(8)))  _Float16 v8h;
typedef __attribute__((ext_vector_type(8)))  float    v8f;
typedef int v4i_ __attribute__((vector_size(16)));   // matches builtin param spelling

// Problem constants (fixed by the reference)
#define B_   4
#define T_   2048
#define D_   1024
#define H_   16
#define G_   4
#define HD_  64
#define KV_  (G_*HD_)      // 256
#define M_   (B_*T_)      // 8192

// 1/sqrt(64) * log2(e): folded into the Q projection so the softmax hot loop
// works directly in the exp2 domain with no extra multiplies.
#define QSCALE_ 0.1803368801111204f

// --------------------------------------------------------------------------
// Async global->LDS (CDNA5, ASYNCcnt) with compile-safe fallback
// --------------------------------------------------------------------------
#if defined(__has_builtin)
#if __has_builtin(__builtin_amdgcn_global_load_async_to_lds_b128)
#define ASYNC_G2L 1
#endif
#endif
#ifndef ASYNC_G2L
#define ASYNC_G2L 0
#endif

__device__ inline void g2l_b128(const _Float16* g, _Float16* l) {
#if ASYNC_G2L
  __builtin_amdgcn_global_load_async_to_lds_b128(
      (__attribute__((address_space(1))) v4i_*)(uintptr_t)g,
      (__attribute__((address_space(3))) v4i_*)(uint32_t)(uintptr_t)l,
      0, 0);
#else
  *(v8h*)l = *(const v8h*)g;
#endif
}
__device__ inline void g2l_drain() {
#if ASYNC_G2L
  asm volatile("s_wait_asynccnt 0" ::: "memory");
#endif
}

// --------------------------------------------------------------------------
// 16-lane all-reduce max using DPP16 only (no LDS): xor1, xor2 via quad_perm,
// then half-row / row mirrors (xor-equivalent once subgroups are uniform).
// --------------------------------------------------------------------------
template <int CTRL>
__device__ inline float dpp_fmax_step(float x) {
  int s = __builtin_bit_cast(int, x);
  int y = __builtin_amdgcn_update_dpp(s, s, CTRL, 0xf, 0xf, true);
  return fmaxf(x, __builtin_bit_cast(float, y));
}
__device__ inline float rowmax16(float x) {
  x = dpp_fmax_step<0xB1>(x);    // quad_perm [1,0,3,2]  (xor 1)
  x = dpp_fmax_step<0x4E>(x);    // quad_perm [2,3,0,1]  (xor 2)
  x = dpp_fmax_step<0x141>(x);   // row_half_mirror      (xor 4 equiv)
  x = dpp_fmax_step<0x140>(x);   // row_mirror           (xor 8 equiv)
  return x;
}

// ---------------------------------------------------------------------------
// Elementwise converts
// ---------------------------------------------------------------------------
__global__ void cvt_f32_to_f16(const float* __restrict__ in,
                               _Float16* __restrict__ out, int n) {
  int i = blockIdx.x * blockDim.x + threadIdx.x;
  if (i < n) out[i] = (_Float16)in[i];
}

// W[K][N] f32  ->  Wt[N][K] f16
__global__ void transpose_cvt(const float* __restrict__ in,
                              _Float16* __restrict__ out, int K, int N) {
  int i = blockIdx.x * blockDim.x + threadIdx.x;
  if (i < K * N) {
    int k = i / N, n = i - k * N;
    out[(size_t)n * K + k] = (_Float16)in[i];
  }
}

// ---------------------------------------------------------------------------
// WMMA GEMM:  C[M,N] = A[M,K](f16,row-major) * Bt[N,K](f16,row-major)^T + bias
// Wave computes a 16x64 strip (4 C-frags); block = 8 waves -> 128x64 tile.
// MODE: 1 = Q scatter [B,H,T,64] (scaled by QSCALE_)   2 = K scatter [B,G,T,64]
//       3 = V^T scatter [B,G,64,T]                     4 = f32 row-major output
// ---------------------------------------------------------------------------
template <int MODE>
__global__ void __launch_bounds__(256)
gemm16(const _Float16* __restrict__ A, const _Float16* __restrict__ Bt,
       const float* __restrict__ bias, void* __restrict__ outp,
       int M, int N, int K) {
  const int lane = threadIdx.x & 31;
  const int wave = threadIdx.x >> 5;
  const int g    = lane >> 4;     // half-wave (0/1)
  const int ln   = lane & 15;

  const int n0 = blockIdx.x * 64;
  const int m0 = blockIdx.y * 128 + wave * 16;

  v8f acc[4] = {};

  const _Float16* arow = A + (size_t)(m0 + ln) * K;

  for (int k0 = 0; k0 < K; k0 += 32) {
    __builtin_prefetch(arow + k0 + 256, 0, 0);          // global_prefetch_b8
    // A-frag 16x32: K = g*16+i (contiguous, matches B)
    v8h alo = *(const v8h*)(arow + k0 + g * 16);
    v8h ahi = *(const v8h*)(arow + k0 + g * 16 + 8);
    v16h a;
#pragma unroll
    for (int i = 0; i < 8; ++i) { a[i] = alo[i]; a[i + 8] = ahi[i]; }

#pragma unroll
    for (int nt = 0; nt < 4; ++nt) {
      // B-frag 32x16: lane n = ln, halves i -> K = g*16+i (row of Bt, contiguous)
      const _Float16* bp = Bt + (size_t)(n0 + nt * 16 + ln) * K + k0 + g * 16;
      v16h b = *(const v16h*)bp;
      acc[nt] = __builtin_amdgcn_wmma_f32_16x16x32_f16(
          false, a, false, b, (short)0, acc[nt], false, false);
    }
  }

  // Epilogue: C layout -> VGPR r holds (M = r+8g, N = ln)
#pragma unroll
  for (int nt = 0; nt < 4; ++nt) {
#pragma unroll
    for (int r = 0; r < 8; ++r) {
      const int mg = m0 + r + 8 * g;
      const int ng = n0 + nt * 16 + ln;
      float v = acc[nt][r] + bias[ng];
      if (MODE == 1) {                 // Q -> [B,H,T,64], pre-scaled
        v *= QSCALE_;
        int b = mg >> 11, t = mg & (T_ - 1);
        int h = ng >> 6, d = ng & 63;
        ((_Float16*)outp)[(((size_t)b * H_ + h) * T_ + t) * HD_ + d] = (_Float16)v;
      } else if (MODE == 2) {          // K -> [B,G,T,64]
        int b = mg >> 11, t = mg & (T_ - 1);
        int h = ng >> 6, d = ng & 63;
        ((_Float16*)outp)[(((size_t)b * G_ + h) * T_ + t) * HD_ + d] = (_Float16)v;
      } else if (MODE == 3) {          // V^T -> [B,G,64,T]
        int b = mg >> 11, t = mg & (T_ - 1);
        int h = ng >> 6, d = ng & 63;
        ((_Float16*)outp)[(((size_t)b * G_ + h) * HD_ + d) * T_ + t] = (_Float16)v;
      } else {                         // f32 row-major
        ((float*)outp)[(size_t)mg * N + ng] = v;
      }
    }
  }
}

// ---------------------------------------------------------------------------
// Fused flash attention (non-causal). Block = 128 threads (4 waves), each wave
// owns 16 query rows. K/V tiles staged cooperatively in LDS, double-buffered,
// via async global->LDS. V tile carries a ones row (row 64) so a 5th WMMA
// accumulates the softmax denominator l alongside O (no cross-lane sums).
// Q:[B,H,T,64] (pre-scaled by QSCALE_)  K:[B,G,T,64]  Vt:[B,G,64,T]
// ctx:[B,T,1024] f16
// ---------------------------------------------------------------------------
__global__ void __launch_bounds__(128)
attn_flash(const _Float16* __restrict__ Qh, const _Float16* __restrict__ Kh,
           const _Float16* __restrict__ VhT, _Float16* __restrict__ ctx) {
  __shared__ _Float16 kbuf[2][32][72];   // 32 keys x 64 dim (+pad) x2 buffers
  __shared__ _Float16 vbuf[2][80][40];   // 80 dim-rows (64 V + 16 ones-ext) x 32 keys (+pad)
  __shared__ _Float16 pbuf[4][16][40];   // per-wave 16x32 P staging (+pad)

  const int tid  = threadIdx.x;
  const int lane = tid & 31;
  const int wave = tid >> 5;
  const int g    = lane >> 4;
  const int ln   = lane & 15;

  const int bh = blockIdx.y;                 // b*H + h
  const int b  = bh >> 4, h = bh & 15;
  const int gr = h >> 2;                     // h / HPG
  const int q0 = blockIdx.x * 64 + wave * 16;

  const _Float16* Qbase = Qh  + ((size_t)bh * T_ + q0) * HD_;
  const _Float16* Kbase = Kh  + (size_t)(b * G_ + gr) * T_ * HD_;
  const _Float16* Vbase = VhT + (size_t)(b * G_ + gr) * HD_ * T_;

  // ones-extension rows of the V tile (row 64 = 1.0, rows 65..79 = 0), both buffers
  for (int i = tid; i < 2 * 16 * 40; i += 128) {
    int pb = i / 640, rem = i - pb * 640;
    int r = rem / 40, c = rem - r * 40;
    vbuf[pb][64 + r][c] = (r == 0) ? (_Float16)1.0f : (_Float16)0.0f;
  }

  // cooperative tile copy: K 32x64 (256 x 16B), V rows 0..63 x 32 (256 x 16B)
  auto issue_copy = [&](int t0, int pb) {
#pragma unroll
    for (int j = 0; j < 2; ++j) {
      int s  = tid + j * 128;
      int kr = s >> 3, kc = (s & 7) * 8;
      g2l_b128(Kbase + (size_t)(t0 + kr) * HD_ + kc, &kbuf[pb][kr][kc]);
      int vr = s >> 2, vc = (s & 3) * 8;
      g2l_b128(Vbase + (size_t)vr * T_ + t0 + vc, &vbuf[pb][vr][vc]);
    }
  };

  // Q A-frags for contraction K=0..31 and K=32..63
  v16h qa[2];
  const _Float16* qrow = Qbase + (size_t)ln * HD_;
#pragma unroll
  for (int kk = 0; kk < 2; ++kk) {
    v8h lo = *(const v8h*)(qrow + kk * 32 + g * 16);
    v8h hi = *(const v8h*)(qrow + kk * 32 + g * 16 + 8);
#pragma unroll
    for (int i = 0; i < 8; ++i) { qa[kk][i] = lo[i]; qa[kk][i + 8] = hi[i]; }
  }

  v8f o[5] = {};            // o[0..3]: 16x64 output; o[4] col 0: running l
  float mrow[8], alpha[8], p0v[8], p1v[8];
#pragma unroll
  for (int r = 0; r < 8; ++r) mrow[r] = -1e30f;

  issue_copy(0, 0);

  constexpr int NT = T_ / 32;
  for (int it = 0; it < NT; ++it) {
    const int pb = it & 1;
    g2l_drain();              // own async copies into buf pb complete
    __syncthreads();          // everyone's copies visible; prev compute done
    if (it + 1 < NT) issue_copy((it + 1) * 32, pb ^ 1);

    // S[16 x 32] = Qs * K^T  (two C-frags, two k-steps each) from LDS
    v8f s2[2] = {};
#pragma unroll
    for (int tt = 0; tt < 2; ++tt) {
#pragma unroll
      for (int kk = 0; kk < 2; ++kk) {
        const _Float16* bp = &kbuf[pb][tt * 16 + ln][kk * 32 + g * 16];
        v8h blo = *(const v8h*)bp;
        v8h bhi = *(const v8h*)(bp + 8);
        v16h bf;
#pragma unroll
        for (int i = 0; i < 8; ++i) { bf[i] = blo[i]; bf[i + 8] = bhi[i]; }
        s2[tt] = __builtin_amdgcn_wmma_f32_16x16x32_f16(
            false, qa[kk], false, bf, (short)0, s2[tt], false, false);
      }
    }

    // Online softmax in exp2 domain; row max via DPP (no LDS traffic)
#pragma unroll
    for (int r = 0; r < 8; ++r) {
      float v0 = s2[0][r], v1 = s2[1][r];
      float mx = rowmax16(fmaxf(v0, v1));
      float mn = fmaxf(mrow[r], mx);
      alpha[r] = exp2f(mrow[r] - mn);
      mrow[r]  = mn;
      p0v[r]   = exp2f(v0 - mn);
      p1v[r]   = exp2f(v1 - mn);
    }
#pragma unroll
    for (int dt = 0; dt < 5; ++dt)
#pragma unroll
      for (int r = 0; r < 8; ++r) o[dt][r] *= alpha[r];

    // Stage P (C layout) -> LDS -> reload as A-frag (same-wave, DS in order)
#pragma unroll
    for (int r = 0; r < 8; ++r) {
      pbuf[wave][r + 8 * g][ln]      = (_Float16)p0v[r];
      pbuf[wave][r + 8 * g][16 + ln] = (_Float16)p1v[r];
    }
    asm volatile("s_wait_dscnt 0" ::: "memory");
    v16h pa;
    {
      const _Float16* prow = &pbuf[wave][ln][0];
      v8h lo = *(const v8h*)(prow + g * 16);
      v8h hi = *(const v8h*)(prow + g * 16 + 8);
#pragma unroll
      for (int i = 0; i < 8; ++i) { pa[i] = lo[i]; pa[i + 8] = hi[i]; }
    }

    // O[0..3] += P*V ; O[4] += P*ones (running softmax denominator)
#pragma unroll
    for (int dt = 0; dt < 5; ++dt) {
      const _Float16* vp = &vbuf[pb][dt * 16 + ln][g * 16];
      v8h blo = *(const v8h*)vp;
      v8h bhi = *(const v8h*)(vp + 8);
      v16h bf;
#pragma unroll
      for (int i = 0; i < 8; ++i) { bf[i] = blo[i]; bf[i + 8] = bhi[i]; }
      o[dt] = __builtin_amdgcn_wmma_f32_16x16x32_f16(
          false, pa, false, bf, (short)0, o[dt], false, false);
    }
  }

  // l lives in o[4] column N=0, i.e. lane g*16 of each half-wave
  float lsum[8];
#pragma unroll
  for (int r = 0; r < 8; ++r) lsum[r] = __shfl(o[4][r], lane & 16, 32);

  // ctx[b][t][h*64 + d] = O / l   (C layout: VGPR r -> row r+8g, col ln)
#pragma unroll
  for (int dt = 0; dt < 4; ++dt)
#pragma unroll
    for (int r = 0; r < 8; ++r) {
      int row = r + 8 * g;
      float val = o[dt][r] / lsum[r];
      ctx[((size_t)(b * T_ + q0 + row)) * D_ + h * HD_ + dt * 16 + ln] =
          (_Float16)val;
    }
}

// ---------------------------------------------------------------------------
extern "C" void kernel_launch(void* const* d_in, const int* in_sizes, int n_in,
                              void* d_out, int out_size, void* d_ws, size_t ws_size,
                              hipStream_t stream) {
  (void)in_sizes; (void)n_in; (void)out_size; (void)ws_size;
  const float* x  = (const float*)d_in[0];
  const float* Wq = (const float*)d_in[1];
  const float* bq = (const float*)d_in[2];
  const float* Wk = (const float*)d_in[3];
  const float* bk = (const float*)d_in[4];
  const float* Wv = (const float*)d_in[5];
  const float* bv = (const float*)d_in[6];
  const float* Wo = (const float*)d_in[7];
  const float* bo = (const float*)d_in[8];
  float* out = (float*)d_out;

  char* ws = (char*)d_ws;
  _Float16* xh  = (_Float16*)ws; ws += (size_t)M_ * D_ * 2;   // x f16
  _Float16* WqT = (_Float16*)ws; ws += (size_t)D_ * D_ * 2;   // Wq^T
  _Float16* WkT = (_Float16*)ws; ws += (size_t)KV_ * D_ * 2;  // Wk^T
  _Float16* WvT = (_Float16*)ws; ws += (size_t)KV_ * D_ * 2;  // Wv^T
  _Float16* WoT = (_Float16*)ws; ws += (size_t)D_ * D_ * 2;   // Wo^T
  _Float16* Qhd = (_Float16*)ws; ws += (size_t)M_ * D_ * 2;   // [B,H,T,64]
  _Float16* Khd = (_Float16*)ws; ws += (size_t)M_ * KV_ * 2;  // [B,G,T,64]
  _Float16* VtT = (_Float16*)ws; ws += (size_t)M_ * KV_ * 2;  // [B,G,64,T]
  _Float16* ctx = (_Float16*)ws; ws += (size_t)M_ * D_ * 2;   // [B,T,1024]

  // 1) converts / weight transposes
  {
    int n = M_ * D_;
    cvt_f32_to_f16<<<(n + 255) / 256, 256, 0, stream>>>(x, xh, n);
  }
  transpose_cvt<<<(D_ * D_ + 255) / 256, 256, 0, stream>>>(Wq, WqT, D_, D_);
  transpose_cvt<<<(D_ * KV_ + 255) / 256, 256, 0, stream>>>(Wk, WkT, D_, KV_);
  transpose_cvt<<<(D_ * KV_ + 255) / 256, 256, 0, stream>>>(Wv, WvT, D_, KV_);
  transpose_cvt<<<(D_ * D_ + 255) / 256, 256, 0, stream>>>(Wo, WoT, D_, D_);

  // 2) projections (WMMA)
  gemm16<1><<<dim3(D_ / 64,  M_ / 128), 256, 0, stream>>>(xh, WqT, bq, Qhd, M_, D_,  D_);
  gemm16<2><<<dim3(KV_ / 64, M_ / 128), 256, 0, stream>>>(xh, WkT, bk, Khd, M_, KV_, D_);
  gemm16<3><<<dim3(KV_ / 64, M_ / 128), 256, 0, stream>>>(xh, WvT, bv, VtT, M_, KV_, D_);

  // 3) fused flash attention (WMMA + async LDS staging)
  attn_flash<<<dim3(T_ / 64, B_ * H_), 128, 0, stream>>>(Qhd, Khd, VtT, ctx);

  // 4) output projection (WMMA, f32 out)
  gemm16<4><<<dim3(D_ / 64, M_ / 128), 256, 0, stream>>>(ctx, WoT, bo, out, M_, D_, D_);
}
